// MHSA_28157805593114
// MI455X (gfx1250) — matrix operand order repaired and checked by reading, not compiled, not run on hardware
//
#include <hip/hip_runtime.h>

// ---------------- problem constants ----------------
#define DIM      512
#define NTOK     16384      // B*M = 8*2048
#define XB_STRIDE 520       // bf16 row stride for LN output (bank spread)
#define QKV_STRIDE 576      // 64 * 9 : Q/K [i][f] view, f-stride padded 8->9
#define V_STRIDE  528       // 8 * 66 : V [f][i] view, i-stride padded 64->66
#define ATT_STRIDE 66       // 64 + 2 pad (even -> 8B-aligned b64 loads)

typedef __attribute__((ext_vector_type(16))) __bf16 v16bf;
typedef __attribute__((ext_vector_type(8)))  __bf16 v8bf;
typedef __attribute__((ext_vector_type(8)))  float  v8f;
typedef __attribute__((ext_vector_type(4)))  float  v4f;
typedef __attribute__((ext_vector_type(2)))  float  v2f;

#define CAT16(lo, hi) __builtin_shufflevector(lo, hi, 0,1,2,3,4,5,6,7,8,9,10,11,12,13,14,15)

// ------------- weight prep: fp32 [k][n] -> bf16 [mat][n][k] -------------
__global__ __launch_bounds__(256) void mhsa_prep_weights(
    const float* __restrict__ Wq, const float* __restrict__ Wk,
    const float* __restrict__ Wv, unsigned short* __restrict__ wt_raw) {
  __bf16* wt = (__bf16*)wt_raw;
  int idx = blockIdx.x * 256 + threadIdx.x;     // [0, 3*512*512)
  int mat = idx >> 18;                          // 512*512 = 2^18
  int rem = idx & 262143;
  int n   = rem >> 9;
  int k   = rem & 511;
  const float* W = (mat == 0) ? Wq : ((mat == 1) ? Wk : Wv);
  wt[idx] = (__bf16)W[k * DIM + n];
}

// ------------- fused LN + QKV GEMM + per-token attention -------------
__global__ __launch_bounds__(256) void mhsa_fused(
    const float* __restrict__ inp,
    const unsigned short* __restrict__ wt_raw,
    const float* __restrict__ bq, const float* __restrict__ bk,
    const float* __restrict__ bv,
    const float* __restrict__ gamma, const float* __restrict__ beta,
    float* __restrict__ out) {
  const __bf16* wt = (const __bf16*)wt_raw;

  extern __shared__ char smem_raw[];
  __bf16* xb    = (__bf16*)smem_raw;                        // 16*520 bf16
  float*  qkQK  = (float*)(smem_raw + 16 * XB_STRIDE * 2);  // 2*16*576 f32 (Q,K)
  float*  vbuf  = qkQK + 2 * 16 * QKV_STRIDE;               // 16*528 f32 (V, [f][i])
  float*  attS  = vbuf + 16 * V_STRIDE;                     // 8*16*66 f32 (per-wave)
  float*  lnred = attS;                                     // aliased (phase 0 only)

  const int tid  = threadIdx.x;
  const int wave = tid >> 5;
  const int lane = tid & 31;
  const int half = lane >> 4;
  const int l16  = lane & 15;
  const int wg   = blockIdx.x;

  // ---------- Phase 0: LayerNorm -> bf16 in LDS ----------
  {
    const int tokl = tid >> 4;        // 16 tokens, 16 lanes each
    const int j16  = tid & 15;        // 32 contiguous floats per lane
    const float* xin = inp + (long)(wg * 16 + tokl) * DIM + j16 * 32;
    float xv[32];
    float s = 0.f, s2 = 0.f;
#pragma unroll
    for (int u = 0; u < 8; ++u) {
      v4f t = *(const v4f*)(xin + u * 4);
#pragma unroll
      for (int q = 0; q < 4; ++q) {
        float x = t[q];
        xv[u * 4 + q] = x;
        s += x; s2 += x * x;
      }
    }
    lnred[tid] = s; lnred[256 + tid] = s2;
    __syncthreads();
    float ts = 0.f, ts2 = 0.f;
#pragma unroll
    for (int u = 0; u < 16; ++u) {
      ts  += lnred[tokl * 16 + u];
      ts2 += lnred[256 + tokl * 16 + u];
    }
    const float mu   = ts * (1.0f / DIM);
    const float var  = ts2 * (1.0f / DIM) - mu * mu;
    const float rstd = rsqrtf(var + 1e-5f);
#pragma unroll
    for (int u = 0; u < 32; ++u) {
      int d = j16 * 32 + u;
      float xn = (xv[u] - mu) * rstd * gamma[d] + beta[d];
      xb[tokl * XB_STRIDE + d] = (__bf16)xn;
    }
    __syncthreads();
  }

  // ---------- Phase 1: QKV = x @ W + b via bf16 WMMA ----------
  // 48 pair-jobs = {Q,K,V} x 16 pairs of adjacent 16-col tiles.
  // One A load feeds two independent WMMA chains; K-loop fully unrolled so the
  // scheduler pipelines loads without loop-carried register rotation.
  const int koffA = half ? 16 : 0;   // A lanes>=16 hold K 16..31
  const int koffB = half ? 16 : 0;   // B lanes>=16 hold K 16..31
  const __bf16* xrow = xb + l16 * XB_STRIDE;

  for (int pr = wave; pr < 48; pr += 8) {
    const int mat = pr >> 4;                    // pr / 16
    const int n0  = (pr & 15) << 5;             // 32-column pair
    const int nA  = n0 + l16;
    const int nB  = n0 + 16 + l16;
    const __bf16* wt0 = wt + ((long)mat << 18) + (long)nA * DIM;
    const __bf16* wt1 = wt + ((long)mat << 18) + (long)nB * DIM;
    __builtin_prefetch(wt0, 0, 3);              // global_prefetch_b8 (L2-resident)
    __builtin_prefetch(wt1, 0, 3);

    v8f c0 = {}, c1 = {};
#pragma unroll
    for (int kk = 0; kk < 16; ++kk) {
      const int ka = kk * 32 + koffA;
      const int kb = kk * 32 + koffB;
      v8bf alo  = *(const v8bf*)(xrow + ka);
      v8bf ahi  = *(const v8bf*)(xrow + ka + 8);
      v8bf b0lo = *(const v8bf*)(wt0 + kb);
      v8bf b0hi = *(const v8bf*)(wt0 + kb + 8);
      v8bf b1lo = *(const v8bf*)(wt1 + kb);
      v8bf b1hi = *(const v8bf*)(wt1 + kb + 8);
      v16bf a = CAT16(alo, ahi);
      c0 = __builtin_amdgcn_wmma_f32_16x16x32_bf16(false, a, false, CAT16(b0lo, b0hi),
                                                   (short)0, c0, false, false);
      c1 = __builtin_amdgcn_wmma_f32_16x16x32_bf16(false, a, false, CAT16(b1lo, b1hi),
                                                   (short)0, c1, false, false);
    }
    const float* bp = (mat == 0) ? bq : ((mat == 1) ? bk : bv);
    const float bnA = bp[nA];
    const float bnB = bp[nB];
    if (mat == 2) {                             // V: store [f][i] transposed view
      float* dstA = vbuf + ((nA & 7) * ATT_STRIDE + (nA >> 3));
      float* dstB = vbuf + ((nB & 7) * ATT_STRIDE + (nB >> 3));
#pragma unroll
      for (int r = 0; r < 8; ++r) {
        const int m = r + half * 8;             // C layout: row = token
        dstA[m * V_STRIDE] = c0[r] + bnA;
        dstB[m * V_STRIDE] = c1[r] + bnB;
      }
    } else {                                    // Q (pre-scaled by 1/8) and K
      const float scl = (mat == 0) ? 0.125f : 1.0f;
      float* dstA = qkQK + mat * (16 * QKV_STRIDE) + ((nA >> 3) * 9 + (nA & 7));
      float* dstB = qkQK + mat * (16 * QKV_STRIDE) + ((nB >> 3) * 9 + (nB & 7));
#pragma unroll
      for (int r = 0; r < 8; ++r) {
        const int m = r + half * 8;
        dstA[m * QKV_STRIDE] = (c0[r] + bnA) * scl;
        dstB[m * QKV_STRIDE] = (c1[r] + bnB) * scl;
      }
    }
  }
  __syncthreads();

  // ---------- Phase 2: per-token 64x8 attention via f32 WMMA ----------
  for (int tt = 0; tt < 2; ++tt) {
    const int tok = wave + tt * 8;
    const float* Qb = qkQK + tok * QKV_STRIDE;
    const float* Kb = qkQK + 16 * QKV_STRIDE + tok * QKV_STRIDE;
    const float* Vb = vbuf + tok * V_STRIDE;
    float* as = attS + wave * (16 * ATT_STRIDE);
    const long tokbase = (long)(wg * 16 + tok) * DIM;
    const int fclamp = (l16 < 8) ? l16 : 0;     // keep V reads in-bounds, no EXEC change

    for (int ti = 0; ti < 4; ++ti) {
      v8f e[4];
      // scores tile row-block ti: S = (Q/8)(64x8) @ K^T(8x64); K-dim 8 = 2x K=4
#pragma unroll
      for (int tj = 0; tj < 4; ++tj) {
        v8f c = {};
        const int mrow = ti * 16 + l16;
        const int nrow = tj * 16 + l16;
#pragma unroll
        for (int ks = 0; ks < 2; ++ks) {
          const int k0 = ks * 4 + half * 2;   // f32 A/B layout: lanes>=16 hold K+2
          v2f a, b;
          a.x = Qb[mrow * 9 + k0];
          a.y = Qb[mrow * 9 + k0 + 1];
          b.x = Kb[nrow * 9 + k0];
          b.y = Kb[nrow * 9 + k0 + 1];
          c = __builtin_amdgcn_wmma_f32_16x16x4_f32(false, a, false, b,
                                                    (short)0, c, false, false);
        }
        e[tj] = c;
      }
      // softmax over the 64 columns: row (r+half*8) lives on one 16-lane group
#pragma unroll
      for (int r = 0; r < 8; ++r) {
        float mx = fmaxf(fmaxf(e[0][r], e[1][r]), fmaxf(e[2][r], e[3][r]));
#pragma unroll
        for (int off = 8; off >= 1; off >>= 1)
          mx = fmaxf(mx, __shfl_xor(mx, off, 32));
        float sum = 0.f;
#pragma unroll
        for (int tj = 0; tj < 4; ++tj) {
          float v = __expf(e[tj][r] - mx);
          e[tj][r] = v;
          sum += v;
        }
#pragma unroll
        for (int off = 8; off >= 1; off >>= 1)
          sum += __shfl_xor(sum, off, 32);
        const float rcp = 1.0f / sum;
#pragma unroll
        for (int tj = 0; tj < 4; ++tj) e[tj][r] *= rcp;
      }
      // spill att block (C layout) to LDS so it can be reloaded in A layout
#pragma unroll
      for (int r = 0; r < 8; ++r)
#pragma unroll
        for (int tj = 0; tj < 4; ++tj)
          as[(r + half * 8) * ATT_STRIDE + tj * 16 + l16] = e[tj][r];
      __syncthreads();
      // out_block(16x8) = att(16x64) @ V(64x8): 16 K=4 steps, N cols 8..15 zero
      v8f o = {};
#pragma unroll 4
      for (int ks = 0; ks < 16; ++ks) {
        const int k0 = ks * 4 + half * 2;
        v2f a  = *(const v2f*)(as + l16 * ATT_STRIDE + k0);       // aligned b64
        v2f bl = *(const v2f*)(Vb + fclamp * ATT_STRIDE + k0);    // V[f][k0..k0+1]
        v2f b;
        b.x = (l16 < 8) ? bl.x : 0.f;
        b.y = (l16 < 8) ? bl.y : 0.f;
        o = __builtin_amdgcn_wmma_f32_16x16x4_f32(false, a, false, b,
                                                  (short)0, o, false, false);
      }
      // residual add + store: out dim d = i*8 + f
      if (l16 < 8) {
#pragma unroll
        for (int r = 0; r < 8; ++r) {
          const int i = ti * 16 + r + half * 8;
          const long d = tokbase + i * 8 + l16;
          out[d] = inp[d] + o[r];
        }
      }
      __syncthreads();
    }
  }
}

extern "C" void kernel_launch(void* const* d_in, const int* in_sizes, int n_in,
                              void* d_out, int out_size, void* d_ws, size_t ws_size,
                              hipStream_t stream) {
  (void)in_sizes; (void)n_in; (void)out_size; (void)ws_size;
  const float* inp   = (const float*)d_in[0];
  const float* Wq    = (const float*)d_in[1];
  const float* bq    = (const float*)d_in[2];
  const float* Wk    = (const float*)d_in[3];
  const float* bk    = (const float*)d_in[4];
  const float* Wv    = (const float*)d_in[5];
  const float* bv    = (const float*)d_in[6];
  const float* gamma = (const float*)d_in[7];
  const float* beta  = (const float*)d_in[8];
  float* out = (float*)d_out;
  unsigned short* wt = (unsigned short*)d_ws;   // 3*512*512 bf16 = 1.5 MB

  mhsa_prep_weights<<<(3 * 512 * 512) / 256, 256, 0, stream>>>(Wq, Wk, Wv, wt);

  const size_t smem = 16 * XB_STRIDE * 2                   // xb (bf16)
                    + 2 * 16 * QKV_STRIDE * sizeof(float)  // Q,K
                    + 16 * V_STRIDE * sizeof(float)        // V ([f][i])
                    + 8 * 16 * ATT_STRIDE * sizeof(float); // att scratch (+LN alias)
  mhsa_fused<<<NTOK / 16, 256, smem, stream>>>(inp, wt, bq, bk, bv,
                                               gamma, beta, out);
}
